// TokenSuppression_29205777613501
// MI455X (gfx1250) — matrix operand order repaired and checked by reading, not compiled, not run on hardware
//
#include <hip/hip_runtime.h>
#include <math.h>

// ---- problem constants ----
#define SEQn 577
#define Bn   64
#define Dn   768
#define Hn   12
#define Kn   8
#define ROWS ((SEQn - 1) * Bn)       // 36864 patch rows, r = p*Bn + b
#define TILES (ROWS / 16)            // 2304
#define ATT_STRIDE ((size_t)SEQn * SEQn)  // 332929
#define EPSv 1e-8f

typedef __attribute__((ext_vector_type(16))) _Float16 v16h;
typedef __attribute__((ext_vector_type(8)))  float    v8f;

// ---------- kernel 0: zero the atomic accumulators (ws is not re-poisoned) ----------
__global__ void zero_kernel(float* scal) {
    if (threadIdx.x < 4) scal[threadIdx.x] = 0.0f;
}

// ---------- kernel 1: per-batch fg mask + text norms ----------
__global__ __launch_bounds__(256) void fg_kernel(const float* __restrict__ attn,
                                                 const float* __restrict__ tc,
                                                 float* __restrict__ fg,
                                                 float* __restrict__ tnorm) {
    __shared__ float ca[576];
    __shared__ float red[256];
    __shared__ float stats[2];
    const int b = blockIdx.x, tid = threadIdx.x;

    for (int p = tid; p < 576; p += 256) {
        float s = 0.0f;
        for (int hh = 0; hh < Hn; ++hh)
            s += attn[(size_t)(b * Hn + hh) * ATT_STRIDE + 1 + p];
        ca[p] = s * (1.0f / (float)Hn);
    }
    __syncthreads();

    float s1 = 0.0f, s2 = 0.0f;
    for (int p = tid; p < 576; p += 256) { float v = ca[p]; s1 += v; s2 += v * v; }

    red[tid] = s1; __syncthreads();
    for (int off = 128; off > 0; off >>= 1) { if (tid < off) red[tid] += red[tid + off]; __syncthreads(); }
    if (tid == 0) stats[0] = red[0];
    __syncthreads();
    red[tid] = s2; __syncthreads();
    for (int off = 128; off > 0; off >>= 1) { if (tid < off) red[tid] += red[tid + off]; __syncthreads(); }
    if (tid == 0) stats[1] = red[0];
    __syncthreads();

    const float mu  = stats[0] / 576.0f;
    const float var = fmaxf(stats[1] - 576.0f * mu * mu, 0.0f) / 575.0f;
    const float cut = mu + 1.5f * sqrtf(var);
    for (int p = tid; p < 576; p += 256)
        fg[(size_t)p * Bn + b] = (ca[p] > cut) ? 1.0f : 0.0f;
    __syncthreads();

    float tn = 0.0f;
    for (int d = tid; d < Dn; d += 256) { float v = tc[(size_t)b * Dn + d]; tn += v * v; }
    red[tid] = tn; __syncthreads();
    for (int off = 128; off > 0; off >>= 1) { if (tid < off) red[tid] += red[tid + off]; __syncthreads(); }
    if (tid == 0) tnorm[b] = sqrtf(red[0]);
}

// ---------- kernel 2: pass 1 — norms/dot/anomaly stats + WMMA rank-8 coefficients ----------
// One wave handles one 16-row tile. A: 16x32 f16 frag; B: pc^T zero-padded to 16 cols.
__global__ __launch_bounds__(256) void p1_kernel(const float* __restrict__ x,
                                                 const float* __restrict__ tcls,
                                                 const float* __restrict__ pc,
                                                 const float* __restrict__ fg,
                                                 const float* __restrict__ tnorm,
                                                 float* __restrict__ anomaly,
                                                 float* __restrict__ coeff,
                                                 float* scal) {
    const int lane = threadIdx.x & 31;
    const int wid  = threadIdx.x >> 5;
    const int tile = blockIdx.x * 8 + wid;
    const int h    = lane >> 4;        // lane half (0/1)
    const int m    = lane & 15;        // row within tile (A), column N (B/C)
    const int row  = tile * 16 + m;
    const int brow = row & (Bn - 1);   // row % 64 == batch index

    const float* Arow = x + (size_t)(Bn + row) * Dn;   // patch row (p,b) contiguous
    const float* Trow = tcls + (size_t)brow * Dn;
    const float  bmask = (m < Kn) ? 1.0f : 0.0f;
    const float* Brow  = pc + (size_t)((m < Kn) ? m : 0) * Dn;

    v8f acc = {};
    float nrm2 = 0.0f, dotv = 0.0f;

    for (int k0 = 0; k0 < Dn; k0 += 32) {
        const int ka = k0 + 8 * h;
        float4 a0 = *(const float4*)(Arow + ka);
        float4 a1 = *(const float4*)(Arow + ka + 4);
        float4 a2 = *(const float4*)(Arow + ka + 16);
        float4 a3 = *(const float4*)(Arow + ka + 20);
        float4 t0 = *(const float4*)(Trow + ka);
        float4 t1 = *(const float4*)(Trow + ka + 4);
        float4 t2 = *(const float4*)(Trow + ka + 16);
        float4 t3 = *(const float4*)(Trow + ka + 20);

        nrm2 += a0.x*a0.x + a0.y*a0.y + a0.z*a0.z + a0.w*a0.w
              + a1.x*a1.x + a1.y*a1.y + a1.z*a1.z + a1.w*a1.w
              + a2.x*a2.x + a2.y*a2.y + a2.z*a2.z + a2.w*a2.w
              + a3.x*a3.x + a3.y*a3.y + a3.z*a3.z + a3.w*a3.w;
        dotv += a0.x*t0.x + a0.y*t0.y + a0.z*t0.z + a0.w*t0.w
              + a1.x*t1.x + a1.y*t1.y + a1.z*t1.z + a1.w*t1.w
              + a2.x*t2.x + a2.y*t2.y + a2.z*t2.z + a2.w*t2.w
              + a3.x*t3.x + a3.y*t3.y + a3.z*t3.z + a3.w*t3.w;

        v16h af;
        af[0]=(_Float16)a0.x; af[1]=(_Float16)a0.y; af[2]=(_Float16)a0.z; af[3]=(_Float16)a0.w;
        af[4]=(_Float16)a1.x; af[5]=(_Float16)a1.y; af[6]=(_Float16)a1.z; af[7]=(_Float16)a1.w;
        af[8]=(_Float16)a2.x; af[9]=(_Float16)a2.y; af[10]=(_Float16)a2.z; af[11]=(_Float16)a2.w;
        af[12]=(_Float16)a3.x; af[13]=(_Float16)a3.y; af[14]=(_Float16)a3.z; af[15]=(_Float16)a3.w;

        const int kb = k0 + 8 * h;
        float4 b0 = *(const float4*)(Brow + kb);
        float4 b1 = *(const float4*)(Brow + kb + 4);
        float4 b2 = *(const float4*)(Brow + kb + 16);
        float4 b3 = *(const float4*)(Brow + kb + 20);
        v16h bf;
        bf[0]=(_Float16)(b0.x*bmask); bf[1]=(_Float16)(b0.y*bmask); bf[2]=(_Float16)(b0.z*bmask); bf[3]=(_Float16)(b0.w*bmask);
        bf[4]=(_Float16)(b1.x*bmask); bf[5]=(_Float16)(b1.y*bmask); bf[6]=(_Float16)(b1.z*bmask); bf[7]=(_Float16)(b1.w*bmask);
        bf[8]=(_Float16)(b2.x*bmask); bf[9]=(_Float16)(b2.y*bmask); bf[10]=(_Float16)(b2.z*bmask); bf[11]=(_Float16)(b2.w*bmask);
        bf[12]=(_Float16)(b3.x*bmask); bf[13]=(_Float16)(b3.y*bmask); bf[14]=(_Float16)(b3.z*bmask); bf[15]=(_Float16)(b3.w*bmask);

        acc = __builtin_amdgcn_wmma_f32_16x16x32_f16(false, af, false, bf,
                                                     (short)0, acc, false, false);
    }

    // combine the two lane-halves (each covered half the K columns of row m)
    nrm2 += __shfl_xor(nrm2, 16, 32);
    dotv += __shfl_xor(dotv, 16, 32);

    if (h == 0) {
        float nrm = sqrtf(nrm2);
        float tn  = tnorm[brow];
        float sim = dotv / (fmaxf(nrm, EPSv) * fmaxf(tn, EPSv));
        float a   = nrm * (1.0f - fmaxf(sim, 0.0f));
        anomaly[row] = a;
        if (fg[row] < 0.5f) {
            atomicAdd(&scal[0], 1.0f);
            atomicAdd(&scal[1], a);
            atomicAdd(&scal[2], a * a);
        }
        atomicMax((int*)&scal[3], __float_as_int(a));  // anomaly >= 0
    }

    if (m < Kn) {
        #pragma unroll
        for (int r = 0; r < 8; ++r)
            coeff[(size_t)(tile * 16 + r + 8 * h) * Kn + m] = acc[r];
    }
}

// ---------- kernel 3: pass 2 — gate + rank-8 reconstruction + blend ----------
__global__ __launch_bounds__(256) void p2_kernel(const float* __restrict__ x,
                                                 const float* __restrict__ pc,
                                                 const float* __restrict__ fg,
                                                 const float* __restrict__ anomaly,
                                                 const float* __restrict__ coeff,
                                                 const float* __restrict__ scal,
                                                 float* __restrict__ out) {
    const int lane = threadIdx.x & 31;
    const int row  = blockIdx.x * 8 + (threadIdx.x >> 5);

    const float n    = scal[0];
    const float s1   = scal[1];
    const float s2   = scal[2];
    const float amax = scal[3];
    const float mu   = s1 / fmaxf(n, 1.0f);
    const float var  = fmaxf(s2 - n * mu * mu, 0.0f) / fmaxf(n - 1.0f, 1.0f);
    const float thr  = (n > 0.0f) ? (mu + 3.0f * sqrtf(var)) : amax;

    const float a    = anomaly[row];
    const float fgv  = fg[row];
    float gate = 1.0f / (1.0f + expf(-5.0f * (thr - a)));
    gate = fgv + (1.0f - fgv) * gate;
    const float og = 1.0f - gate;

    float c[Kn];
    #pragma unroll
    for (int k = 0; k < Kn; ++k) c[k] = coeff[(size_t)row * Kn + k];

    const float* Xr = x   + (size_t)(Bn + row) * Dn;
    float*       Or = out + (size_t)(Bn + row) * Dn;

    #pragma unroll
    for (int it = 0; it < 6; ++it) {
        const int d0 = it * 128 + lane * 4;
        float4 xv = *(const float4*)(Xr + d0);
        float px = 0.0f, py = 0.0f, pz = 0.0f, pw = 0.0f;
        #pragma unroll
        for (int k = 0; k < Kn; ++k) {
            float4 pv = *(const float4*)(pc + (size_t)k * Dn + d0);
            px += c[k] * pv.x; py += c[k] * pv.y;
            pz += c[k] * pv.z; pw += c[k] * pv.w;
        }
        float4 ov;
        ov.x = gate * xv.x + og * px;
        ov.y = gate * xv.y + og * py;
        ov.z = gate * xv.z + og * pz;
        ov.w = gate * xv.w + og * pw;
        *(float4*)(Or + d0) = ov;
    }
}

// ---------- kernel 4: CLS row pass-through ----------
__global__ void cls_kernel(const float* __restrict__ x, float* __restrict__ out) {
    const int i = blockIdx.x * 256 + threadIdx.x;   // float4 index
    if (i < (Bn * Dn) / 4)
        ((float4*)out)[i] = ((const float4*)x)[i];
}

extern "C" void kernel_launch(void* const* d_in, const int* in_sizes, int n_in,
                              void* d_out, int out_size, void* d_ws, size_t ws_size,
                              hipStream_t stream) {
    const float* x    = (const float*)d_in[0];
    const float* attn = (const float*)d_in[1];
    const float* tcls = (const float*)d_in[2];
    const float* pc   = (const float*)d_in[3];
    float* out = (float*)d_out;

    // workspace layout (floats)
    float* ws      = (float*)d_ws;
    float* fg      = ws;                        // ROWS
    float* tnorm   = fg + ROWS;                 // Bn
    float* anomaly = tnorm + Bn;                // ROWS
    float* coeff   = anomaly + ROWS;            // ROWS * Kn
    float* scal    = coeff + (size_t)ROWS * Kn; // 4: n, sum, sumsq, amax

    zero_kernel<<<1, 32, 0, stream>>>(scal);
    fg_kernel<<<Bn, 256, 0, stream>>>(attn, tcls, fg, tnorm);
    cls_kernel<<<(Bn * Dn / 4 + 255) / 256, 256, 0, stream>>>(x, out);
    p1_kernel<<<TILES / 8, 256, 0, stream>>>(x, tcls, pc, fg, tnorm, anomaly, coeff, scal);
    p2_kernel<<<ROWS / 8, 256, 0, stream>>>(x, pc, fg, anomaly, coeff, scal, out);
}
